// ConvVQVAE_4080218931433
// MI455X (gfx1250) — matrix operand order repaired and checked by reading, not compiled, not run on hardware
//
#include <hip/hip_runtime.h>
#include <hip/hip_bf16.h>

// ---------------------------------------------------------------------------
// CDNA5 (gfx1250) VQ-VAE forward: implicit-GEMM convs on v_wmma_f32_16x16x32_bf16
// + TDM (tensor_load_to_lds) staging of the VQ codebook into LDS.
// Per K-step each lane's 16 B-operand taps are one 4x4 (conv) / 2x2 (deconv)
// patch of a single input channel -> vectorized patch loads on interior pixels.
// ---------------------------------------------------------------------------

typedef __attribute__((ext_vector_type(16))) __bf16 v16bf;
typedef __attribute__((ext_vector_type(8)))  float  v8f;
typedef __attribute__((ext_vector_type(4)))  unsigned int v4u;
typedef __attribute__((ext_vector_type(8)))  int v8i;
typedef __attribute__((ext_vector_type(4)))  int v4i;

#define BATCH 128

#if defined(__has_builtin)
#  if __has_builtin(__builtin_amdgcn_tensor_load_to_lds) && __has_builtin(__builtin_amdgcn_s_wait_tensorcnt)
#    define HAVE_TDM 1
#  endif
#endif

// fp32 -> bf16, round-to-nearest-even
__device__ __forceinline__ unsigned short f2bf(float f) {
  union { float f; unsigned u; } x; x.f = f;
  unsigned u = x.u;
  u += 0x7FFFu + ((u >> 16) & 1u);
  return (unsigned short)(u >> 16);
}

union BV16 {            // type-pun 16 bf16 <-> 2x 16B loads <-> builtin operand
  v16bf v;
  unsigned short u[16];
  uint4 q[2];
};

// --------------------------- weight packing --------------------------------
// A matrix row-major [Mpad][Kpad] bf16, K index = ci*16 + ky*4 + kx.
// flip=1: transposed-conv weight [Cin,Cout,4,4] with spatial flip baked in.
__global__ void pack_weights(const float* __restrict__ w, unsigned short* __restrict__ apack,
                             int Cout, int Cin, int Mpad, int Kpad, int flip) {
  int i = blockIdx.x * blockDim.x + threadIdx.x;
  int tot = Mpad * Kpad;
  if (i >= tot) return;
  int m = i / Kpad, k = i - m * Kpad;
  int ci = k >> 4, r = k & 15, ky = r >> 2, kx = r & 3;
  float v = 0.f;
  if (m < Cout && ci < Cin) {
    if (!flip) v = w[((m * Cin + ci) * 4 + ky) * 4 + kx];
    else       v = w[((ci * Cout + m) * 4 + (3 - ky)) * 4 + (3 - kx)];
  }
  apack[i] = f2bf(v);
}

// codebook -> bf16 [e][c] (B-operand friendly), per-code squared norms, zero loss
__global__ void pack_codebook(const float* __restrict__ cb, unsigned short* __restrict__ cbp,
                              float* __restrict__ cbn2, float* __restrict__ loss) {
  int e = blockIdx.x * blockDim.x + threadIdx.x;
  if (e == 0) *loss = 0.f;
  if (e >= 512) return;
  float s = 0.f;
  for (int c = 0; c < 64; ++c) {
    float v = cb[e * 64 + c];
    cbp[e * 64 + c] = f2bf(v);
    s += v * v;
  }
  cbn2[e] = s;
}

// --------------------- implicit-GEMM conv / deconv -------------------------
// GEMM: D[M=Cout][N=B*Hout*Wout] = A[M][K=Cin*16] x B[K][N], K-step 32 (bf16 WMMA).
// One wave gathers each B tile ONCE and accumulates MT m-tiles (MT*16 >= Cout).
// DEC=0: stride-2 pad-1 conv. DEC=1: transposed conv as 2x-dilated conv, pad-top PT.
// ACT: 0 none, 1 relu, 2 sigmoid.
template<int Cin, int H, int W, int Cout, int Hout, int Wout, int DEC, int PT, int ACT, int MT>
__global__ void __launch_bounds__(128)
conv_wmma(const float* __restrict__ in, const unsigned short* __restrict__ apack,
          const float* __restrict__ bias, float* __restrict__ out)
{
  constexpr int Kpad = ((Cin * 16 + 31) / 32) * 32;
  constexpr long long Ntot = (long long)BATCH * Hout * Wout;
  const int lane = threadIdx.x & 31;
  const int wave = threadIdx.x >> 5;
  const int half = lane >> 4;
  const int l16  = lane & 15;
  const long long ntile = (long long)blockIdx.x * 4 + wave;

  const long long n = ntile * 16 + l16;         // this lane's output pixel (B-col / C-col)
  const bool nvalid = (n < Ntot);
  const int hw  = Hout * Wout;
  const int b   = (int)(n / hw);
  const int rem = (int)(n - (long long)b * hw);
  const int oy  = rem / Wout;
  const int ox  = rem - oy * Wout;

  v8f acc[MT];
  {
    v8f z = {};
#pragma unroll
    for (int mt = 0; mt < MT; ++mt) acc[mt] = z;
  }

  for (int kk = 0; kk < Kpad; kk += 32) {
    // ---- B operand: one 4x4 (or 2x2) patch of channel ci per lane per K-step ----
    const int ci = (kk >> 4) + half;
    const bool lvalid = nvalid && (ci < Cin);
    const float* base = in + ((size_t)b * Cin + ci) * (size_t)(H * W);
    BV16 bu;
    if (DEC == 0) {
      const int iy0 = oy * 2 - 1, ix0 = ox * 2 - 1;
      if (lvalid && iy0 >= 0 && ix0 >= 0 && iy0 + 3 < H && ix0 + 3 < W) {
        // interior: 4 row loads of 16B (unaligned-capable global_load_b128)
#pragma unroll
        for (int ky = 0; ky < 4; ++ky) {
          float4 row;
          __builtin_memcpy(&row, base + (size_t)(iy0 + ky) * W + ix0, 16);
          bu.u[ky * 4 + 0] = f2bf(row.x);
          bu.u[ky * 4 + 1] = f2bf(row.y);
          bu.u[ky * 4 + 2] = f2bf(row.z);
          bu.u[ky * 4 + 3] = f2bf(row.w);
        }
      } else {
#pragma unroll
        for (int t = 0; t < 16; ++t) {
          const int ky = t >> 2, kx = t & 3;
          const int iy = iy0 + ky, ix = ix0 + kx;
          float v = 0.f;
          if (lvalid && (unsigned)iy < (unsigned)H && (unsigned)ix < (unsigned)W)
            v = base[(size_t)iy * W + ix];
          bu.u[t] = f2bf(v);
        }
      }
    } else {
      // transposed conv: taps with even dilated coords form a 2x2 input patch
      const int kyp = (oy + PT) & 1, kxp = (ox + PT) & 1;
      const int iy0 = (oy + kyp - PT) >> 1;       // even numerator, arithmetic shift
      const int ix0 = (ox + kxp - PT) >> 1;
      if (lvalid && iy0 >= 0 && ix0 >= 0 && iy0 + 1 < H && ix0 + 1 < W) {
        float2 r0, r1;
        __builtin_memcpy(&r0, base + (size_t)iy0 * W + ix0, 8);
        __builtin_memcpy(&r1, base + (size_t)(iy0 + 1) * W + ix0, 8);
        const float p[2][2] = { { r0.x, r0.y }, { r1.x, r1.y } };
#pragma unroll
        for (int t = 0; t < 16; ++t) {
          const int ky = t >> 2, kx = t & 3;
          float v = 0.f;
          if (((ky & 1) == kyp) && ((kx & 1) == kxp))
            v = p[(ky - kyp) >> 1][(kx - kxp) >> 1];
          bu.u[t] = f2bf(v);
        }
      } else {
#pragma unroll
        for (int t = 0; t < 16; ++t) {
          const int ky = t >> 2, kx = t & 3;
          const int dy = oy + ky - PT, dx = ox + kx - PT;
          float v = 0.f;
          if (lvalid && ((dy | dx) >= 0) && !((dy | dx) & 1)) {
            const int iy = dy >> 1, ix = dx >> 1;
            if (iy < H && ix < W) v = base[(size_t)iy * W + ix];
          }
          bu.u[t] = f2bf(v);
        }
      }
    }
    // ---- A operands (2 contiguous 16B loads per m-tile), MT WMMAs per B tile ----
#pragma unroll
    for (int mt = 0; mt < MT; ++mt) {
      const unsigned short* ap = apack + (size_t)(mt * 16 + l16) * Kpad + kk + half * 16;
      BV16 au;
      au.q[0] = *(const uint4*)(ap);            // K = kk + half*16 + [0..7]
      au.q[1] = *(const uint4*)(ap + 8);        // K = kk + half*16 + 8 + [0..7]
      __builtin_prefetch(ap + 32, 0, 3);        // global_prefetch_b8, next K-step
      acc[mt] = __builtin_amdgcn_wmma_f32_16x16x32_bf16(false, au.v, false, bu.v,
                                                        (short)0, acc[mt], false, false);
    }
  }

  // ---- epilogue: C layout lane holds rows r + 8*half, col l16 ----
#pragma unroll
  for (int mt = 0; mt < MT; ++mt) {
#pragma unroll
    for (int r = 0; r < 8; ++r) {
      const int oc = mt * 16 + r + 8 * half;
      if (nvalid && oc < Cout) {
        float v = acc[mt][r] + bias[oc];
        if (ACT == 1) v = fmaxf(v, 0.f);
        if (ACT == 2) v = 1.f / (1.f + __expf(-v));
        out[(((size_t)b * Cout + oc) * Hout + oy) * Wout + ox] = v;
      }
    }
  }
}

// ------------------------- vector quantization -----------------------------
// Per wave: 16 z-vectors x 512 codes, dim 64. dist = |z|^2 + |e|^2 - 2 z.e
// The packed bf16 codebook (64KB) is staged into LDS once per workgroup via the
// Tensor Data Mover (tensor_load_to_lds, TENSORcnt), then B operands come from
// ds_load_b128. Cross-lane argmin via shfl_xor (ds_bpermute).
__global__ void __launch_bounds__(128)
vq_wmma(const float* __restrict__ z, const unsigned short* __restrict__ cbp,
        const float* __restrict__ cb, const float* __restrict__ cbn2,
        float* __restrict__ zq, float* __restrict__ idx_out, float* __restrict__ loss)
{
  __shared__ uint4 smem4[4096];                 // 64KB: packed codebook [512][64] bf16

  const int lane = threadIdx.x & 31;
  const int wave = threadIdx.x >> 5;
  const int half = lane >> 4;
  const int l16  = lane & 15;

#if defined(HAVE_TDM)
  if (wave == 0) {
    // ---- TDM descriptor (ISA 8.3/8.4): 1 row of 32768 bf16 elements -> LDS ----
    const unsigned long long ga = (unsigned long long)(uintptr_t)cbp;
    v4u g0;
    g0[0] = 1u;                                       // count=1, user mode
    g0[1] = (unsigned)(uintptr_t)(&smem4[0]);         // lds_addr (low 32 bits = LDS offset)
    g0[2] = (unsigned)(ga & 0xFFFFFFFFu);             // global_addr[31:0]
    g0[3] = (unsigned)((ga >> 32) & 0x01FFFFFFu) | 0x80000000u;  // addr[56:32] | type=2
    v8i g1;
    g1[0] = (int)(1u << 16);       // workgroup_mask=0, data_size=1 (2 bytes)
    g1[1] = (int)0x80000000u;      // tensor_dim0[15:0]=0x8000 in bits[31:16]
    g1[2] = (int)0x00010000u;      // tensor_dim0[31:16]=0, tensor_dim1=1
    g1[3] = (int)0x80000000u;      // tile_dim0=0x8000 in bits[31:16]
    g1[4] = 1;                     // tile_dim1=1, tile_dim2=0
    g1[5] = 32768;                 // tensor_dim0_stride[31:0]
    g1[6] = 0;                     // stride hi / tensor_dim1_stride lo
    g1[7] = 0;
    v4i gz = {0, 0, 0, 0};
#  if __has_include(<hip/amd_detail/amd_gfx1250_TDM.h>)
    v8i gz8 = {0, 0, 0, 0, 0, 0, 0, 0};
    __builtin_amdgcn_tensor_load_to_lds(g0, g1, gz, gz, gz8, 0);   // therock 6-arg
#  else
    __builtin_amdgcn_tensor_load_to_lds(g0, g1, gz, gz, 0);        // ROCm 7.2 5-arg
#  endif
    __builtin_amdgcn_s_wait_tensorcnt(0);
  }
  __syncthreads();
#else
  {  // fallback: cooperative copy global -> LDS
    const uint4* src = (const uint4*)cbp;
    for (int i = threadIdx.x; i < 4096; i += 128) smem4[i] = src[i];
    __syncthreads();
  }
#endif

  const int ntile = blockIdx.x * 4 + wave;      // 2048 tiles of 16 vectors
  const int n = ntile * 16 + l16;
  const int b = n >> 8, rem = n & 255, hy = rem >> 4, hx = rem & 15;
  const int zoff = (b << 14) | (hy << 4) | hx;  // z[((b*64+c)*16+hy)*16+hx], c stride 256

  // |z|^2 for this lane's vector
  float zn2 = 0.f;
  for (int c = 0; c < 64; ++c) { float v = z[zoff + (c << 8)]; zn2 += v * v; }
  float zn2m[8];
#pragma unroll
  for (int r = 0; r < 8; ++r) zn2m[r] = __shfl(zn2, r + 8 * half, 32);

  // A operand (z-vectors, K=64 in two 32-steps), contiguous K order, k = 16*half + t
  BV16 a0, a1;
#pragma unroll
  for (int t = 0; t < 16; ++t) {
    const int k0 = half * 16 + t;                     // {0..15} / {16..31}
    a0.u[t] = f2bf(z[zoff + (k0 << 8)]);
    a1.u[t] = f2bf(z[zoff + ((k0 + 32) << 8)]);
  }

  float best[8]; int bidx[8];
#pragma unroll
  for (int r = 0; r < 8; ++r) { best[r] = 3.4e38f; bidx[r] = 0; }

  for (int et = 0; et < 32; ++et) {
    const int e = et * 16 + l16;
    const int bo = e * 8 + half * 2;             // uint4 index into LDS codebook
    BV16 b0, b1;
    b0.q[0] = smem4[bo];      b0.q[1] = smem4[bo + 1];     // ds_load_b128
    b1.q[0] = smem4[bo + 4];  b1.q[1] = smem4[bo + 5];
    v8f g = {};
    g = __builtin_amdgcn_wmma_f32_16x16x32_bf16(false, a0.v, false, b0.v, (short)0, g, false, false);
    g = __builtin_amdgcn_wmma_f32_16x16x32_bf16(false, a1.v, false, b1.v, (short)0, g, false, false);
    const float ce = cbn2[e];
#pragma unroll
    for (int r = 0; r < 8; ++r) {
      const float d = zn2m[r] + ce - 2.f * g[r];
      if (d < best[r]) { best[r] = d; bidx[r] = e; }
    }
  }

  // argmin across the 16 lanes of each half (same m, distinct e); lowest-index tiebreak
#pragma unroll
  for (int r = 0; r < 8; ++r) {
#pragma unroll
    for (int off = 1; off < 16; off <<= 1) {
      const float od = __shfl_xor(best[r], off, 32);
      const int   oi = __shfl_xor(bidx[r], off, 32);
      if (od < best[r] || (od == best[r] && oi < bidx[r])) { best[r] = od; bidx[r] = oi; }
    }
  }

  if (l16 == 0) {                 // lanes 0 & 16 write results for m = r + 8*half
    float lsum = 0.f;
    const int ntbase = ntile * 16;
#pragma unroll 1
    for (int r = 0; r < 8; ++r) {
      const int nn = ntbase + r + 8 * half;
      const int e  = bidx[r];
      idx_out[nn] = (float)e;
      const int bb = nn >> 8, rr = nn & 255, yy = rr >> 4, xx = rr & 15;
      const int zo = (bb << 14) | (yy << 4) | xx;
      for (int c = 0; c < 64; ++c) {
        const float qc = cb[e * 64 + c];
        const float zc = z[zo + (c << 8)];
        const float d  = qc - zc;
        lsum += d * d;
        zq[zo + (c << 8)] = qc;   // straight-through forward value == q
      }
    }
    atomicAdd(loss, lsum);
  }
}

__global__ void vq_finalize(const float* __restrict__ loss, float* __restrict__ out_loss) {
  *out_loss = 1.25f * (*loss) / (32768.f * 64.f);   // (1 + 0.25) * mean((q-z)^2)
}

// ------------------------------ launcher -----------------------------------
extern "C" void kernel_launch(void* const* d_in, const int* in_sizes, int n_in,
                              void* d_out, int out_size, void* d_ws, size_t ws_size,
                              hipStream_t stream) {
  const float* x      = (const float*)d_in[0];
  const float* enc_w1 = (const float*)d_in[1];  const float* enc_b1 = (const float*)d_in[2];
  const float* enc_w2 = (const float*)d_in[3];  const float* enc_b2 = (const float*)d_in[4];
  const float* enc_w3 = (const float*)d_in[5];  const float* enc_b3 = (const float*)d_in[6];
  const float* codebook = (const float*)d_in[7];
  const float* dec_w1 = (const float*)d_in[8];  const float* dec_b1 = (const float*)d_in[9];
  const float* dec_w2 = (const float*)d_in[10]; const float* dec_b2 = (const float*)d_in[11];
  const float* dec_w3 = (const float*)d_in[12]; const float* dec_b3 = (const float*)d_in[13];
  float* out = (float*)d_out;

  // workspace layout (float offsets)
  float* ws = (float*)d_ws;
  const size_t O_H1   = 0;                         // 128*32*64*64  (h1, reused for d2 out)
  const size_t O_H2   = O_H1 + 16777216;           // 128*64*32*32  (h2, reused for d1 out)
  const size_t O_Z    = O_H2 + 8388608;            // 128*64*16*16
  const size_t O_ZQ   = O_Z  + 2097152;
  const size_t O_CBN  = O_ZQ + 2097152;            // 512 code norms
  const size_t O_LOSS = O_CBN + 512;               // 1 (+pad to keep 16B alignment below)
  unsigned short* us  = (unsigned short*)(ws + O_LOSS + 8);   // bf16 region, 16B aligned
  unsigned short* W1 = us;            // 32*64
  unsigned short* W2 = us + 2048;     // 64*512
  unsigned short* W3 = us + 34816;    // 64*1024
  unsigned short* D1 = us + 100352;   // 64*1024
  unsigned short* D2 = us + 165888;   // 32*1024
  unsigned short* D3 = us + 198656;   // 16*512
  unsigned short* CB = us + 206848;   // 512*64

  // d_out = x_recon[128*3*131*131] | vq_loss | idx[32768]
  const size_t XR = 6589824;
  float* out_loss = out + XR;
  float* out_idx  = out + XR + 1;

  // ---- pack weights (bf16 GEMM A-matrices) & codebook ----
  pack_weights<<<(32 * 64 + 255) / 256,   256, 0, stream>>>(enc_w1, W1, 32, 3, 32, 64, 0);
  pack_weights<<<(64 * 512 + 255) / 256,  256, 0, stream>>>(enc_w2, W2, 64, 32, 64, 512, 0);
  pack_weights<<<(64 * 1024 + 255) / 256, 256, 0, stream>>>(enc_w3, W3, 64, 64, 64, 1024, 0);
  pack_weights<<<(64 * 1024 + 255) / 256, 256, 0, stream>>>(dec_w1, D1, 64, 64, 64, 1024, 1);
  pack_weights<<<(32 * 1024 + 255) / 256, 256, 0, stream>>>(dec_w2, D2, 32, 64, 32, 1024, 1);
  pack_weights<<<(16 * 512 + 255) / 256,  256, 0, stream>>>(dec_w3, D3, 3, 32, 16, 512, 1);
  pack_codebook<<<2, 256, 0, stream>>>(codebook, CB, ws + O_CBN, ws + O_LOSS);

  // ---- encoder (MT m-tiles per wave: B gathered once, reused across all Cout) ----
  conv_wmma<3, 128, 128, 32, 64, 64, 0, 0, 1, 2><<<8192, 128, 0, stream>>>(x,         W1, enc_b1, ws + O_H1);
  conv_wmma<32, 64, 64, 64, 32, 32, 0, 0, 1, 4><<<2048, 128, 0, stream>>>(ws + O_H1, W2, enc_b2, ws + O_H2);
  conv_wmma<64, 32, 32, 64, 16, 16, 0, 0, 0, 4><<<512,  128, 0, stream>>>(ws + O_H2, W3, enc_b3, ws + O_Z);

  // ---- vector quantization (idx, z_q, loss partials) ----
  vq_wmma<<<512, 128, 0, stream>>>(ws + O_Z, CB, codebook, ws + O_CBN,
                                   ws + O_ZQ, out_idx, ws + O_LOSS);

  // ---- decoder (transposed convs as dilated convs; buffers ping-pong) ----
  conv_wmma<64, 16, 16, 64, 32, 32, 1, 2, 1, 4><<<2048,  128, 0, stream>>>(ws + O_ZQ, D1, dec_b1, ws + O_H2);
  conv_wmma<64, 32, 32, 32, 64, 64, 1, 2, 1, 2><<<8192,  128, 0, stream>>>(ws + O_H2, D2, dec_b2, ws + O_H1);
  conv_wmma<32, 64, 64, 3, 131, 131, 1, 3, 2, 1><<<34322, 128, 0, stream>>>(ws + O_H1, D3, dec_b3, out);

  vq_finalize<<<1, 1, 0, stream>>>(ws + O_LOSS, out_loss);

  (void)in_sizes; (void)n_in; (void)out_size; (void)ws_size;
}
